// MambaBlock_77180562309781
// MI455X (gfx1250) — matrix operand order repaired and checked by reading, not compiled, not run on hardware
//
#include <hip/hip_runtime.h>
#include <hip/hip_bf16.h>

// ---------------------------------------------------------------------------
// MI455X (gfx1250) Mamba block + FFN.
// GEMMs: v_wmma_f32_16x16x32_bf16 with TDM (tensor_load_to_lds) double-buffered
// LDS staging. Weights pre-transposed to [N,K] bf16 so both operands are plain
// 2D row-major TDM tiles (TDM cannot transpose).
// ---------------------------------------------------------------------------

typedef __bf16 bf16;
typedef __attribute__((ext_vector_type(8)))  __bf16 v8bf;
typedef __attribute__((ext_vector_type(16))) __bf16 v16bf;
typedef __attribute__((ext_vector_type(8)))  float  v8f;
typedef __attribute__((ext_vector_type(4)))  unsigned int u32x4;
typedef __attribute__((ext_vector_type(8)))  int i32x8;
typedef __attribute__((ext_vector_type(4)))  int i32x4;

#define TILE_M 128
#define TILE_N 128
#define TILE_K 32
#define LDS_STRIDE 40            // bf16/row in LDS: 64B data + 16B pad = 80B rows
#define HALF_BUF (TILE_M * LDS_STRIDE)   // one tile (A or B)
#define BUF_STRIDE (2 * HALF_BUF)        // one double-buffer slot (A tile + B tile)

enum { EPI_BIAS = 0, EPI_SIGMOID = 1, EPI_RESID = 2, EPI_GELU_BF16 = 3 };

__device__ __forceinline__ unsigned int lds_off(const void* p) {
  // generic LDS pointer: addr[63:32] is the shared aperture, addr[31:0] is the
  // LDS byte offset (ISA 10.2 aperture mapping) -> truncate.
  return (unsigned int)(uintptr_t)p;
}

// ---------------------------------------------------------------------------
// TDM: async 2D tile load (32 cols x 128 rows of bf16) into LDS with HW row
// padding of 16B per 64B (-> 80B LDS pitch). D# per CDNA5 ISA 8.3/8.4.
// ---------------------------------------------------------------------------
__device__ __forceinline__ void tdm_tile_load(unsigned int lds_addr,
                                              const bf16* gptr,
                                              unsigned int dim0 /*K elems*/,
                                              unsigned int rows /*tensor dim1*/) {
  const unsigned long long ga = (unsigned long long)(uintptr_t)gptr;
  u32x4 g0;
  g0[0] = 1u;                                   // count=1, no gather
  g0[1] = lds_addr;                             // lds_addr [63:32]
  g0[2] = (unsigned int)ga;                     // global_addr [95:64]
  g0[3] = (unsigned int)((ga >> 32) & 0x1FFFFFFu) | (2u << 30);  // addr hi | type=2
  i32x8 g1;
  // data_size=1 (2B), pad_enable, pad_interval=3 (16 DW), pad_amount=3 (4 DW)
  g1[0] = (1 << 16) | (1 << 20) | (3 << 22) | (3 << 25);
  g1[1] = (int)((dim0 & 0xFFFFu) << 16);                    // tensor_dim0[15:0]
  g1[2] = (int)((dim0 >> 16) | ((rows & 0xFFFFu) << 16));   // dim0 hi | dim1 lo
  g1[3] = (int)((rows >> 16) | (TILE_K << 16));             // dim1 hi | tile_dim0
  g1[4] = TILE_M;                                           // tile_dim1 (tile_dim2=0)
  g1[5] = (int)dim0;                                        // tensor_dim0_stride lo
  g1[6] = 0;                                                // stride hi | dim1_stride lo
  g1[7] = 0;
  i32x4 z4; z4[0] = z4[1] = z4[2] = z4[3] = 0;
#if defined(__clang_major__) && (__clang_major__ >= 23)
  i32x8 z8; for (int i = 0; i < 8; ++i) z8[i] = 0;
  __builtin_amdgcn_tensor_load_to_lds(g0, g1, z4, z4, z8, 0);
#else
  __builtin_amdgcn_tensor_load_to_lds(g0, g1, z4, z4, 0);
#endif
}

// ---------------------------------------------------------------------------
// Tiled bf16 WMMA GEMM: out = epi(A[M,K] @ Wt[N,K]^T + bias)
// Block tile 128x128, 8 waves (wave32), each wave 32x64 = 2x4 WMMA tiles.
// TDM double-buffered LDS staging; k-loop body is pure ds_load_b128 + wmma.
// ---------------------------------------------------------------------------
template <int EPI>
__global__ __launch_bounds__(256) void gemm_bf16_wmma(
    const bf16* __restrict__ A, const bf16* __restrict__ Wt,
    const float* __restrict__ bias, const float* __restrict__ resid,
    float* __restrict__ outF, bf16* __restrict__ outB, int M, int N, int K) {
  __shared__ bf16 smem[2 * BUF_STRIDE];  // [buf][Atile | Btile]

  const int tid  = threadIdx.x;
  const int lane = tid & 31;
  const int wave = tid >> 5;
  const int wm   = wave & 3;   // 4 wave rows  -> 32-row strips
  const int wn   = wave >> 2;  // 2 wave cols  -> 64-col strips
  const int m0   = blockIdx.y * TILE_M;
  const int n0   = blockIdx.x * TILE_N;
  const int lmod  = lane & 15;
  const int lhalf = lane >> 4;  // 0: lanes 0-15, 1: lanes 16-31

  v8f acc[2][4];
#pragma unroll
  for (int i = 0; i < 2; ++i)
#pragma unroll
    for (int j = 0; j < 4; ++j)
#pragma unroll
      for (int e = 0; e < 8; ++e) acc[i][j][e] = 0.0f;

  const int nK = K / TILE_K;
  if (wave == 0) {  // TDM issue is per-wave; one wave drives both tiles
    tdm_tile_load(lds_off(&smem[0]), A + (size_t)m0 * K, K, M);
    tdm_tile_load(lds_off(&smem[HALF_BUF]), Wt + (size_t)n0 * K, K, N);
    __builtin_amdgcn_s_wait_tensorcnt(0);
  }
  __syncthreads();

  for (int kt = 0; kt < nK; ++kt) {
    const int cur = kt & 1, nxt = cur ^ 1;
    if (wave == 0 && kt + 1 < nK) {  // overlap next tile DMA with WMMA compute
      tdm_tile_load(lds_off(&smem[nxt * BUF_STRIDE]),
                    A + (size_t)m0 * K + (kt + 1) * TILE_K, K, M);
      tdm_tile_load(lds_off(&smem[nxt * BUF_STRIDE + HALF_BUF]),
                    Wt + (size_t)n0 * K + (kt + 1) * TILE_K, K, N);
    }

    const bf16* At = smem + cur * BUF_STRIDE;
    const bf16* Bt = At + HALF_BUF;
    // A fragments (16x32 bf16: lanes 0-15 hold K[0..7],K[16..23]; 16-31 +8)
    v16bf afrag[2];
#pragma unroll
    for (int tm = 0; tm < 2; ++tm) {
      const int m    = wm * 32 + tm * 16 + lmod;
      const int koff = lhalf * 8;
      const v8bf lo = *reinterpret_cast<const v8bf*>(&At[m * LDS_STRIDE + koff]);
      const v8bf hi = *reinterpret_cast<const v8bf*>(&At[m * LDS_STRIDE + koff + 16]);
#pragma unroll
      for (int e = 0; e < 8; ++e) { afrag[tm][e] = lo[e]; afrag[tm][8 + e] = hi[e]; }
    }
    // B fragments (32x16: lanes 0-15 K[0..7],K[16..23]; 16-31 +8) from Wt rows
#pragma unroll
    for (int tn = 0; tn < 4; ++tn) {
      const int n    = wn * 64 + tn * 16 + lmod;
      const int koff = lhalf * 8;
      const v8bf blo = *reinterpret_cast<const v8bf*>(&Bt[n * LDS_STRIDE + koff]);
      const v8bf bhi = *reinterpret_cast<const v8bf*>(&Bt[n * LDS_STRIDE + koff + 16]);
      v16bf bfrag;
#pragma unroll
      for (int e = 0; e < 8; ++e) { bfrag[e] = blo[e]; bfrag[8 + e] = bhi[e]; }
#pragma unroll
      for (int tm = 0; tm < 2; ++tm) {
        acc[tm][tn] = __builtin_amdgcn_wmma_f32_16x16x32_bf16(
            false, afrag[tm], false, bfrag, (short)0, acc[tm][tn], false, false);
      }
    }

    if (wave == 0 && kt + 1 < nK) __builtin_amdgcn_s_wait_tensorcnt(0);
    __syncthreads();
  }

  // ---- epilogue: C/D layout lanes 0-15 -> M[0..7], 16-31 -> M[8..15] ----
#pragma unroll
  for (int tn = 0; tn < 4; ++tn) {
    const int n  = n0 + wn * 64 + tn * 16 + lmod;
    const float bn = bias[n];
#pragma unroll
    for (int tm = 0; tm < 2; ++tm) {
      const int mb = m0 + wm * 32 + tm * 16 + lhalf * 8;
#pragma unroll
      for (int v = 0; v < 8; ++v) {
        const size_t oi = (size_t)(mb + v) * N + n;
        float val = acc[tm][tn][v] + bn;
        if (EPI == EPI_SIGMOID) {
          val = 1.0f / (1.0f + __expf(-val));
          outF[oi] = val * (0.1f - 0.001f) + 0.001f;  // dt range
        } else if (EPI == EPI_RESID) {
          outF[oi] = val + resid[oi];
        } else if (EPI == EPI_GELU_BF16) {
          outB[oi] = (bf16)(0.5f * val * (1.0f + erff(val * 0.70710678118f)));
        } else {
          outF[oi] = val;
        }
      }
    }
  }
}

// ---------------------------------------------------------------------------
// Weight convert + transpose: in[K,N] f32 -> out[N,K] bf16 (LDS 32x32 tiles)
// ---------------------------------------------------------------------------
__global__ __launch_bounds__(256) void transpose_to_bf16(
    const float* __restrict__ in, bf16* __restrict__ out, int K, int N) {
  __shared__ float tile[32][33];
  const int kb = blockIdx.y * 32, nb = blockIdx.x * 32;
  const int tx = threadIdx.x & 31, ty = threadIdx.x >> 5;  // 32 x 8
#pragma unroll
  for (int r = 0; r < 32; r += 8)
    tile[ty + r][tx] = in[(size_t)(kb + ty + r) * N + nb + tx];
  __syncthreads();
#pragma unroll
  for (int r = 0; r < 32; r += 8)
    out[(size_t)(nb + ty + r) * K + kb + tx] = (bf16)tile[tx][ty + r];
}

// ---------------------------------------------------------------------------
// LayerNorm over 1024-wide rows -> bf16 output (GEMM A operand)
// ---------------------------------------------------------------------------
__global__ __launch_bounds__(256) void layernorm_bf16(
    const float* __restrict__ x, const float* __restrict__ g,
    const float* __restrict__ bta, bf16* __restrict__ out) {
  __shared__ float s1[256], s2[256];
  const int row = blockIdx.x;
  const float* xr = x + (size_t)row * 1024;
  float v[4], lsum = 0.f, lsq = 0.f;
#pragma unroll
  for (int i = 0; i < 4; ++i) {
    v[i] = xr[threadIdx.x + i * 256];
    lsum += v[i]; lsq += v[i] * v[i];
  }
  s1[threadIdx.x] = lsum; s2[threadIdx.x] = lsq;
  __syncthreads();
  for (int st = 128; st > 0; st >>= 1) {
    if ((int)threadIdx.x < st) {
      s1[threadIdx.x] += s1[threadIdx.x + st];
      s2[threadIdx.x] += s2[threadIdx.x + st];
    }
    __syncthreads();
  }
  const float mean = s1[0] * (1.0f / 1024.0f);
  const float var  = s2[0] * (1.0f / 1024.0f) - mean * mean;
  const float rstd = rsqrtf(var + 1e-5f);
#pragma unroll
  for (int i = 0; i < 4; ++i) {
    const int c = threadIdx.x + i * 256;
    out[(size_t)row * 1024 + c] = (bf16)((v[i] - mean) * rstd * g[c] + bta[c]);
  }
}

// ---------------------------------------------------------------------------
// Causal depthwise conv, k=4, over x_proj = xg[:, 0:2048]
// ---------------------------------------------------------------------------
__global__ __launch_bounds__(256) void dwconv_kernel(
    const float* __restrict__ xg, const float* __restrict__ cw,
    const float* __restrict__ cb, float* __restrict__ xcF, bf16* __restrict__ xcB) {
  const int idx = blockIdx.x * 256 + threadIdx.x;  // < 2*1024*2048
  const int d = idx & 2047;
  const int s = (idx >> 11) & 1023;
  const int b = idx >> 21;
  float acc = cb[d];
#pragma unroll
  for (int j = 0; j < 4; ++j) {
    const int sp = s - 3 + j;
    if (sp >= 0) acc += cw[d * 4 + j] * xg[(size_t)(b * 1024 + sp) * 4096 + d];
  }
  const size_t oi = (size_t)(b * 1024 + s) * 2048 + d;
  xcF[oi] = acc;
  xcB[oi] = (bf16)acc;
}

// ---------------------------------------------------------------------------
// Selective scan: one thread per (b, d_inner) channel, 16 states in registers.
// A_bar = exp(delta * A) recomputed on the fly. Fuses +Dp*x_proj and *silu(gate).
// ---------------------------------------------------------------------------
__global__ __launch_bounds__(256) void mamba_scan(
    const float* __restrict__ delta, const float* __restrict__ Bm,
    const float* __restrict__ Cm, const float* __restrict__ xg,
    const float* __restrict__ A_log, const float* __restrict__ Dp,
    bf16* __restrict__ yb) {
  const int gidx = blockIdx.x * 256 + threadIdx.x;  // 0..4095
  const int b = gidx >> 11;
  const int d = gidx & 2047;
  float Arow[16], h[16];
#pragma unroll
  for (int n = 0; n < 16; ++n) {
    Arow[n] = -__expf(A_log[d * 16 + n]);
    h[n] = 0.0f;
  }
  const float Dd = Dp[d];
  for (int s = 0; s < 1024; ++s) {
    const size_t row = (size_t)(b * 1024 + s);
    const float dt = delta[row * 2048 + d];
    const float xp = xg[row * 4096 + d];
    const float gt = xg[row * 4096 + 2048 + d];
    const float* Bp = Bm + (row * 2048 + d) * 16;
    const float* Cp = Cm + (row * 2048 + d) * 16;
    // prefetch next timestep's streams (global_prefetch_b8 hint)
    __builtin_prefetch(Bp + 2048 * 16, 0, 1);
    __builtin_prefetch(Cp + 2048 * 16, 0, 1);
    float y = 0.0f;
#pragma unroll
    for (int n = 0; n < 16; ++n) {
      const float a = __expf(dt * Arow[n]);
      h[n] = a * h[n] + dt * Bp[n] * xp;
      y += Cp[n] * h[n];
    }
    y += Dd * xp;
    const float sg = 1.0f / (1.0f + __expf(-gt));
    yb[row * 2048 + d] = (bf16)(y * gt * sg);
  }
}

// ---------------------------------------------------------------------------
extern "C" void kernel_launch(void* const* d_in, const int* in_sizes, int n_in,
                              void* d_out, int out_size, void* d_ws, size_t ws_size,
                              hipStream_t stream) {
  const float* x      = (const float*)d_in[0];
  const float* ln1_g  = (const float*)d_in[1];
  const float* ln1_b  = (const float*)d_in[2];
  const float* W_in   = (const float*)d_in[3];
  const float* b_in   = (const float*)d_in[4];
  const float* conv_w = (const float*)d_in[5];
  const float* conv_b = (const float*)d_in[6];
  const float* A_log  = (const float*)d_in[7];
  const float* W_B    = (const float*)d_in[8];
  const float* b_B    = (const float*)d_in[9];
  const float* W_C    = (const float*)d_in[10];
  const float* b_C    = (const float*)d_in[11];
  const float* Dp     = (const float*)d_in[12];
  const float* W_dt   = (const float*)d_in[13];
  const float* b_dt   = (const float*)d_in[14];
  const float* W_out  = (const float*)d_in[15];
  const float* b_out  = (const float*)d_in[16];
  const float* ln2_g  = (const float*)d_in[17];
  const float* ln2_b  = (const float*)d_in[18];
  const float* W1     = (const float*)d_in[19];
  const float* b1     = (const float*)d_in[20];
  const float* W2     = (const float*)d_in[21];
  const float* b2     = (const float*)d_in[22];
  float* out = (float*)d_out;

  const int M = 2048;  // B*S
  const int D = 1024, DI = 2048, DFF = 4096;
  const int NBS = DI * 16;  // 32768

  char* ws = (char*)d_ws;
  size_t off = 0;
  auto carve = [&](size_t bytes) -> char* {
    char* p = ws + off;
    off += (bytes + 255) & ~(size_t)255;
    return p;
  };
  // transposed bf16 weights: wt_* is [N, K]
  bf16* wt_in  = (bf16*)carve((size_t)D * 2 * DI * 2);
  bf16* wt_dt  = (bf16*)carve((size_t)DI * DI * 2);
  bf16* wt_B   = (bf16*)carve((size_t)DI * NBS * 2);
  bf16* wt_C   = (bf16*)carve((size_t)DI * NBS * 2);
  bf16* wt_out = (bf16*)carve((size_t)DI * D * 2);
  bf16* wt_1   = (bf16*)carve((size_t)D * DFF * 2);
  bf16* wt_2   = (bf16*)carve((size_t)DFF * D * 2);
  bf16* xn_b   = (bf16*)carve((size_t)M * D * 2);
  float* xg_f  = (float*)carve((size_t)M * 2 * DI * 4);
  float* xc_f  = (float*)carve((size_t)M * DI * 4);
  bf16* xc_b   = (bf16*)carve((size_t)M * DI * 2);
  float* dlt_f = (float*)carve((size_t)M * DI * 4);
  float* Bm_f  = (float*)carve((size_t)M * NBS * 4);
  float* Cm_f  = (float*)carve((size_t)M * NBS * 4);
  bf16* y_b    = (bf16*)carve((size_t)M * DI * 2);
  float* x2_f  = (float*)carve((size_t)M * D * 4);
  bf16* xn2_b  = (bf16*)carve((size_t)M * D * 2);
  bf16* h1_b   = (bf16*)carve((size_t)M * DFF * 2);

  // 1. weights -> bf16, transposed to [N,K] (TDM-friendly row-major tiles)
  transpose_to_bf16<<<dim3(2 * DI / 32, D / 32), 256, 0, stream>>>(W_in, wt_in, D, 2 * DI);
  transpose_to_bf16<<<dim3(DI / 32, DI / 32), 256, 0, stream>>>(W_dt, wt_dt, DI, DI);
  transpose_to_bf16<<<dim3(NBS / 32, DI / 32), 256, 0, stream>>>(W_B, wt_B, DI, NBS);
  transpose_to_bf16<<<dim3(NBS / 32, DI / 32), 256, 0, stream>>>(W_C, wt_C, DI, NBS);
  transpose_to_bf16<<<dim3(D / 32, DI / 32), 256, 0, stream>>>(W_out, wt_out, DI, D);
  transpose_to_bf16<<<dim3(DFF / 32, D / 32), 256, 0, stream>>>(W1, wt_1, D, DFF);
  transpose_to_bf16<<<dim3(D / 32, DFF / 32), 256, 0, stream>>>(W2, wt_2, DFF, D);

  // 2. LN1 -> xn (bf16)
  layernorm_bf16<<<M, 256, 0, stream>>>(x, ln1_g, ln1_b, xn_b);

  // 3. xg = xn @ W_in + b_in  [2048, 4096]
  gemm_bf16_wmma<EPI_BIAS><<<dim3(2 * DI / TILE_N, M / TILE_M), 256, 0, stream>>>(
      xn_b, wt_in, b_in, nullptr, xg_f, nullptr, M, 2 * DI, D);

  // 4. causal depthwise conv -> xc (f32 + bf16)
  dwconv_kernel<<<M * DI / 256, 256, 0, stream>>>(xg_f, conv_w, conv_b, xc_f, xc_b);

  // 5. delta = sigmoid(xc @ W_dt + b_dt)*(dtmax-dtmin)+dtmin
  gemm_bf16_wmma<EPI_SIGMOID><<<dim3(DI / TILE_N, M / TILE_M), 256, 0, stream>>>(
      xc_b, wt_dt, b_dt, nullptr, dlt_f, nullptr, M, DI, DI);

  // 6/7. Bm, Cm = xc @ W_B / W_C  [2048, 32768] — the dominant GEMMs
  gemm_bf16_wmma<EPI_BIAS><<<dim3(NBS / TILE_N, M / TILE_M), 256, 0, stream>>>(
      xc_b, wt_B, b_B, nullptr, Bm_f, nullptr, M, NBS, DI);
  gemm_bf16_wmma<EPI_BIAS><<<dim3(NBS / TILE_N, M / TILE_M), 256, 0, stream>>>(
      xc_b, wt_C, b_C, nullptr, Cm_f, nullptr, M, NBS, DI);

  // 8. selective scan (+Dp*x_proj, *silu(gate)) -> y (bf16)
  mamba_scan<<<16, 256, 0, stream>>>(dlt_f, Bm_f, Cm_f, xg_f, A_log, Dp, y_b);

  // 9. x2 = x + y @ W_out + b_out
  gemm_bf16_wmma<EPI_RESID><<<dim3(D / TILE_N, M / TILE_M), 256, 0, stream>>>(
      y_b, wt_out, b_out, x, x2_f, nullptr, M, D, DI);

  // 10. LN2 -> xn2 (bf16)
  layernorm_bf16<<<M, 256, 0, stream>>>(x2_f, ln2_g, ln2_b, xn2_b);

  // 11. h1 = gelu(xn2 @ W1 + b1) -> bf16
  gemm_bf16_wmma<EPI_GELU_BF16><<<dim3(DFF / TILE_N, M / TILE_M), 256, 0, stream>>>(
      xn2_b, wt_1, b1, nullptr, nullptr, h1_b, M, DFF, D);

  // 12. out = x2 + h1 @ W2 + b2
  gemm_bf16_wmma<EPI_RESID><<<dim3(D / TILE_N, M / TILE_M), 256, 0, stream>>>(
      h1_b, wt_2, b2, x2_f, out, nullptr, M, D, DFF);
}
